// DynamicNetwork_35854386987051
// MI455X (gfx1250) — matrix operand order repaired and checked by reading, not compiled, not run on hardware
//
#include <hip/hip_runtime.h>
#include <cstdint>
#include <cstddef>

// ---- WMMA fragment types (wave32, gfx1250) ----
typedef __attribute__((ext_vector_type(16))) _Float16 v16h;
typedef __attribute__((ext_vector_type(8)))  _Float16 v8h;
typedef __attribute__((ext_vector_type(8)))  float    v8f;
typedef __attribute__((ext_vector_type(2)))  float    f2v;

#define N_PTS    1048576
#define NLVL     16
#define NT_LVL   12         // levels >= this use non-temporal table loads
#define TSZ      524288u
#define TMASK    (TSZ - 1u)
#define FDIM     224        // 204 padded to 7 K-chunks of 32
#define HIDDEN   64
#define PTS_BLK  64
#define THREADS  128        // 4 waves
#define GROWTH   1.3819f
#define PRIME1   2654435761u

#define WMMA_F16(A, B, C) \
  __builtin_amdgcn_wmma_f32_16x16x32_f16(false, (A), false, (B), (short)0, (C), false, false)

// convert 16 consecutive f32 (64B, 16B-aligned) to a B-fragment row of 16 f16
__device__ __forceinline__ v16h cvt_row16(const float* __restrict__ src) {
  const float4* q = (const float4*)src;
  float4 a = q[0], b = q[1], c = q[2], d = q[3];
  v16h r;
  r[0]  = (_Float16)a.x; r[1]  = (_Float16)a.y; r[2]  = (_Float16)a.z; r[3]  = (_Float16)a.w;
  r[4]  = (_Float16)b.x; r[5]  = (_Float16)b.y; r[6]  = (_Float16)b.z; r[7]  = (_Float16)b.w;
  r[8]  = (_Float16)c.x; r[9]  = (_Float16)c.y; r[10] = (_Float16)c.z; r[11] = (_Float16)c.w;
  r[12] = (_Float16)d.x; r[13] = (_Float16)d.y; r[14] = (_Float16)d.z; r[15] = (_Float16)d.w;
  return r;
}

// one bilinear hashgrid level; NT selects non-temporal table loads
template <bool NT>
__device__ __forceinline__ void hg_level(const f2v* __restrict__ tl,
                                         float px, float py, float res,
                                         _Float16* __restrict__ dst) {
  const float fx = px * res, fy = py * res;
  const float flx = floorf(fx), fly = floorf(fy);
  const float wx = fx - flx, wy = fy - fly;
  const uint32_t x0 = (uint32_t)(int)flx, y0 = (uint32_t)(int)fly;
  const uint32_t x1 = x0 + 1u, y1 = y0 + 1u;
  const uint32_t hy0 = y0 * PRIME1, hy1 = y1 * PRIME1;
  f2v v00, v10, v01, v11;
  if (NT) {
    v00 = __builtin_nontemporal_load(&tl[(x0 ^ hy0) & TMASK]);
    v10 = __builtin_nontemporal_load(&tl[(x1 ^ hy0) & TMASK]);
    v01 = __builtin_nontemporal_load(&tl[(x0 ^ hy1) & TMASK]);
    v11 = __builtin_nontemporal_load(&tl[(x1 ^ hy1) & TMASK]);
  } else {
    v00 = tl[(x0 ^ hy0) & TMASK];
    v10 = tl[(x1 ^ hy0) & TMASK];
    v01 = tl[(x0 ^ hy1) & TMASK];
    v11 = tl[(x1 ^ hy1) & TMASK];
  }
  const float w00 = (1.0f - wx) * (1.0f - wy), w10 = wx * (1.0f - wy);
  const float w01 = (1.0f - wx) * wy,          w11 = wx * wy;
  dst[0] = (_Float16)(v00.x * w00 + v10.x * w10 + v01.x * w01 + v11.x * w11);
  dst[1] = (_Float16)(v00.y * w00 + v10.y * w10 + v01.y * w01 + v11.y * w11);
}

__global__ __launch_bounds__(THREADS)
void ngp_fused_kernel(const float* __restrict__ p0, const float* __restrict__ p1,
                      const float* __restrict__ p2, const float* __restrict__ p3,
                      const float* __restrict__ p4, const float* __restrict__ p5,
                      const float* __restrict__ tables,
                      const float* __restrict__ W1, const float* __restrict__ W2,
                      const float* __restrict__ W3, float* __restrict__ out)
{
  // 28KB + 8KB + 8KB + 2KB = 46KB static LDS
  __shared__ __align__(32) _Float16 featLds[PTS_BLK * FDIM];
  __shared__ __align__(32) _Float16 hidLds [PTS_BLK * HIDDEN];
  __shared__ __align__(32) _Float16 w2Lds  [HIDDEN * HIDDEN];
  __shared__ __align__(32) _Float16 w3Lds  [HIDDEN * 16];

  const int t    = threadIdx.x;
  const int blk0 = blockIdx.x * PTS_BLK;

  // ---- stage small weights into LDS as f16 ----
  for (int i = t; i < HIDDEN * HIDDEN; i += THREADS) w2Lds[i] = (_Float16)W2[i];
  for (int i = t; i < HIDDEN * 16; i += THREADS) {
    int r = i >> 4, c = i & 15;
    w3Lds[i] = (_Float16)((c < 3) ? W3[r * 3 + c] : 0.0f);
  }

  // ---- phase 1: hash-grid encoding (2 threads per point, 3 sets each) ----
  {
    const int p    = t >> 1;
    const int half = t & 1;
    const int gp   = blk0 + p;
    _Float16* fb = &featLds[p * FDIM];
    if (half) {
      #pragma unroll
      for (int k = 204; k < FDIM; ++k) fb[k] = (_Float16)0.0f;
    }
    #pragma unroll 1
    for (int ss = 0; ss < 3; ++ss) {
      const int s = half * 3 + ss;
      const float* P = (s == 0) ? p0 : (s == 1) ? p1 : (s == 2) ? p2
                    : (s == 3) ? p3 : (s == 4) ? p4 : p5;
      const float2 pc = ((const float2*)P)[gp];
      fb[192 + 2 * s]     = (_Float16)pc.x;   // raw coords appended per set
      fb[192 + 2 * s + 1] = (_Float16)pc.y;
      const float* tabset = tables + (size_t)s * NLVL * TSZ * 2u;
      float res = 16.0f;
      // low/mid levels: cache-friendly, regular loads
      #pragma unroll 2
      for (int l = 0; l < NT_LVL; ++l) {
        hg_level<false>((const f2v*)(tabset + (size_t)l * TSZ * 2u),
                        pc.x, pc.y, res, &fb[s * 32 + 2 * l]);
        res *= GROWTH;
      }
      // top levels: effectively random over ~4MB tables -> non-temporal
      #pragma unroll 2
      for (int l = NT_LVL; l < NLVL; ++l) {
        hg_level<true>((const f2v*)(tabset + (size_t)l * TSZ * 2u),
                       pc.x, pc.y, res, &fb[s * 32 + 2 * l]);
        res *= GROWTH;
      }
    }
  }
  __syncthreads();

  // ---- phase 2: fused MLP via WMMA; each wave owns 16 points ----
  const int wave = t >> 5;
  const int lane = t & 31;
  const int m    = lane & 15;   // A: row M / D: col N
  const int hi   = lane >> 4;   // lane-half selector

  // A fragment: 16-bit A 16x32 layout — lane holds row m; K = (hi?8:0)+{0..7} and +16..+23
  auto loadA = [&](const _Float16* base, int rstride, int kb) -> v16h {
    const _Float16* q = base + (wave * 16 + m) * rstride + kb + 8 * hi;
    v8h lo = *(const v8h*)q;
    v8h hh = *(const v8h*)(q + 16);
    v16h a;
    #pragma unroll
    for (int i = 0; i < 8; ++i) { a[i] = lo[i]; a[i + 8] = hh[i]; }
    return a;
  };
  // B fragment: lane holds col N = nb+m; K = (hi?8:0)+{0..7} and +16..+23
  auto loadB = [&](const _Float16* Wl, int ncols, int kb, int nb) -> v16h {
    v16h b;
    #pragma unroll
    for (int i = 0; i < 16; ++i)
      b[i] = Wl[(kb + 8 * hi + (i & 7) + ((i >> 3) << 4)) * ncols + nb + m];
    return b;
  };

  // ---- layer 1: [16x224] x [224x64], B converted f32->f16 from global ----
  // A is zero in K=204..223, so padding rows of B only need a safe address:
  // clamp instead of branching (kills exec-mask + zero-fill overhead).
  v8f c0 = {}, c1 = {}, c2 = {}, c3 = {};
  #pragma unroll 1
  for (int kc = 0; kc < 7; ++kc) {
    v16h a = loadA(featLds, FDIM, kc * 32);
    v16h b0, b1, b2, b3;
    #pragma unroll
    for (int s = 0; s < 16; ++s) {
      const int row = min(kc * 32 + 8 * hi + (s & 7) + ((s >> 3) << 4), 203);
      const float* wr = W1 + row * 64 + m;
      b0[s] = (_Float16)wr[0];
      b1[s] = (_Float16)wr[16];
      b2[s] = (_Float16)wr[32];
      b3[s] = (_Float16)wr[48];
    }
    c0 = WMMA_F16(a, b0, c0);
    c1 = WMMA_F16(a, b1, c1);
    c2 = WMMA_F16(a, b2, c2);
    c3 = WMMA_F16(a, b3, c3);
  }
  // ReLU -> f16 hidden activations (D layout: lane n=m, VGPR v -> row v+8*hi)
  #pragma unroll
  for (int v = 0; v < 8; ++v) {
    const int r = (wave * 16 + v + 8 * hi) * HIDDEN + m;
    hidLds[r +  0] = (_Float16)fmaxf(c0[v], 0.0f);
    hidLds[r + 16] = (_Float16)fmaxf(c1[v], 0.0f);
    hidLds[r + 32] = (_Float16)fmaxf(c2[v], 0.0f);
    hidLds[r + 48] = (_Float16)fmaxf(c3[v], 0.0f);
  }

  // ---- layer 2: [16x64] x [64x64] ----
  v8f d0 = {}, d1 = {}, d2 = {}, d3 = {};
  #pragma unroll
  for (int kc = 0; kc < 2; ++kc) {
    v16h a  = loadA(hidLds, HIDDEN, kc * 32);
    v16h b0 = loadB(w2Lds, HIDDEN, kc * 32, 0);
    v16h b1 = loadB(w2Lds, HIDDEN, kc * 32, 16);
    v16h b2 = loadB(w2Lds, HIDDEN, kc * 32, 32);
    v16h b3 = loadB(w2Lds, HIDDEN, kc * 32, 48);
    d0 = WMMA_F16(a, b0, d0);
    d1 = WMMA_F16(a, b1, d1);
    d2 = WMMA_F16(a, b2, d2);
    d3 = WMMA_F16(a, b3, d3);
  }
  // ReLU -> f16; overwrite our wave's rows of hidLds (wave-local, DS in-order)
  #pragma unroll
  for (int v = 0; v < 8; ++v) {
    const int r = (wave * 16 + v + 8 * hi) * HIDDEN + m;
    hidLds[r +  0] = (_Float16)fmaxf(d0[v], 0.0f);
    hidLds[r + 16] = (_Float16)fmaxf(d1[v], 0.0f);
    hidLds[r + 32] = (_Float16)fmaxf(d2[v], 0.0f);
    hidLds[r + 48] = (_Float16)fmaxf(d3[v], 0.0f);
  }

  // ---- layer 3: [16x64] x [64x16] (cols 0..2 valid) ----
  v8f e = {};
  #pragma unroll
  for (int kc = 0; kc < 2; ++kc) {
    v16h a = loadA(hidLds, HIDDEN, kc * 32);
    v16h b = loadB(w3Lds, 16, kc * 32, 0);
    e = WMMA_F16(a, b, e);
  }
  if (m < 3) {
    #pragma unroll
    for (int v = 0; v < 8; ++v) {
      const int row = blk0 + wave * 16 + v + 8 * hi;
      out[row * 3 + m] = e[v];
    }
  }
}

extern "C" void kernel_launch(void* const* d_in, const int* in_sizes, int n_in,
                              void* d_out, int out_size, void* d_ws, size_t ws_size,
                              hipStream_t stream) {
  (void)in_sizes; (void)n_in; (void)out_size; (void)d_ws; (void)ws_size;
  const float* p0 = (const float*)d_in[0];
  const float* p1 = (const float*)d_in[1];
  const float* p2 = (const float*)d_in[2];
  const float* p3 = (const float*)d_in[3];
  const float* p4 = (const float*)d_in[4];
  const float* p5 = (const float*)d_in[5];
  const float* tables = (const float*)d_in[6];
  const float* W1 = (const float*)d_in[7];
  const float* W2 = (const float*)d_in[8];
  const float* W3 = (const float*)d_in[9];
  float* out = (float*)d_out;

  dim3 grid(N_PTS / PTS_BLK);   // 16384 blocks of 64 points
  dim3 block(THREADS);
  ngp_fused_kernel<<<grid, block, 0, stream>>>(p0, p1, p2, p3, p4, p5,
                                               tables, W1, W2, W3, out);
}
